// Attention_84877143703554
// MI455X (gfx1250) — matrix operand order repaired and checked by reading, not compiled, not run on hardware
//
#include <hip/hip_runtime.h>

typedef __attribute__((ext_vector_type(16))) _Float16 v16h;
typedef __attribute__((ext_vector_type(8)))  _Float16 v8h;
typedef __attribute__((ext_vector_type(8)))  float    v8f;

#define B_SZ 2
#define SEQ  2048
#define DIM  1024
#define NH   16
#define DH   64
#define MTOT (B_SZ * SEQ)          // 4096 tokens total
#define NEG_MAX (-3.402823466e38f)
static constexpr float kScale = 0.03125f;   // 1024^-0.5

// LDS B-tile: 64 rows x 32 halves, padded to 48 halves/row (96B, keeps 16B alignment)
#define BROW 48
#define BTILE (64 * BROW)

__device__ inline unsigned lds_off(const void* p) {
    return (unsigned)(unsigned long long)p;   // low 32 bits of LDS aperture = LDS byte offset
}

// ---------------------------------------------------------------- converts
__global__ void f32_to_f16(const float* __restrict__ in, _Float16* __restrict__ out, int n) {
    int i = blockIdx.x * blockDim.x + threadIdx.x;
    int stride = gridDim.x * blockDim.x;
    for (; i < n; i += stride) out[i] = (_Float16)in[i];
}

// ---------------------------------------------------------------- generic NT GEMM
// C[M,N] = scale * (A[M,K] @ B[N,K]^T) + bias, A/B f16 row-major.
// Block: 256 threads = 8 waves; each wave owns a 16x64 C strip.
// B tile (64 x 32 f16) staged in LDS via async-LDS loads, double buffered.
__global__ __launch_bounds__(256)
void gemm_nt(const _Float16* __restrict__ A, const _Float16* __restrict__ Bm,
             int M, int N, int K, int lda, int ldb,
             _Float16* __restrict__ Ch, float* __restrict__ Cf, int ldc,
             float scale, const float* __restrict__ bias) {
    __shared__ __align__(16) _Float16 smemB[2][BTILE];

    const int lane = threadIdx.x & 31;
    const int wave = threadIdx.x >> 5;
    const int m0   = blockIdx.y * 128 + wave * 16;
    const bool active = (m0 < M);
    const int n0  = blockIdx.x * 64;
    const int l15 = lane & 15;
    const int hi  = lane >> 4;       // 0: lanes 0-15, 1: lanes 16-31
    const int ak0 = hi ? 16 : 0;     // A-fragment K sub-offset (halves)
    const int bk0 = hi ? 16 : 0;     // B-fragment K sub-offset (halves)

    // cooperative async B-tile prefetch: thread t -> row t/4, 16B chunk t%4
    const int pr = threadIdx.x >> 2;     // 0..63
    const int pc = threadIdx.x & 3;      // 0..3
    const _Float16* gB = Bm + (size_t)(n0 + pr) * ldb + pc * 8;
    const unsigned ldsB0 = lds_off(&smemB[0][0]) + pr * (BROW * 2) + pc * 16;

    v8f acc[4] = {};
    const _Float16* arow = A + (size_t)((active ? m0 : 0) + l15) * lda;

    // prime buffer 0
    {
        unsigned lo = ldsB0;
        unsigned long long ga = (unsigned long long)(gB);
        asm volatile("global_load_async_to_lds_b128 %0, %1, off" :: "v"(lo), "v"(ga) : "memory");
    }

    int buf = 0;
    for (int kb = 0; kb < K; kb += 32) {
        asm volatile("s_wait_asynccnt 0x0" ::: "memory");
        __syncthreads();

        if (kb + 32 < K) {   // prefetch next k-slab into the other buffer
            unsigned lo = ldsB0 + (buf ^ 1) * (BTILE * 2);
            unsigned long long ga = (unsigned long long)(gB + kb + 32);
            asm volatile("global_load_async_to_lds_b128 %0, %1, off" :: "v"(lo), "v"(ga) : "memory");
        }

        // A fragment (direct, contiguous per lane)
        v8h alo = *(const v8h*)(arow + kb + ak0);
        v8h ahi = *(const v8h*)(arow + kb + 8 + ak0);
        v16h a;
#pragma unroll
        for (int i = 0; i < 8; ++i) { a[i] = alo[i]; a[i + 8] = ahi[i]; }

        // B fragments from LDS
        v16h bf[4];
#pragma unroll
        for (int t = 0; t < 4; ++t)
            bf[t] = *(const v16h*)(&smemB[buf][(t * 16 + l15) * BROW + bk0]);
#pragma unroll
        for (int t = 0; t < 4; ++t)
            acc[t] = __builtin_amdgcn_wmma_f32_16x16x32_f16(
                false, a, false, bf[t], (short)0, acc[t], false, false);

        __syncthreads();     // reads done before this buffer is overwritten
        buf ^= 1;
    }

    if (!active) return;
#pragma unroll
    for (int r = 0; r < 8; ++r) {
        const size_t m = (size_t)(m0 + r + 8 * hi);
#pragma unroll
        for (int t = 0; t < 4; ++t) {
            const int n = n0 + t * 16 + l15;
            float v = acc[t][r] * scale + (bias ? bias[n] : 0.0f);
            if (Ch) Ch[m * ldc + n] = (_Float16)v;
            else    Cf[m * ldc + n] = v;
        }
    }
}

// ---------------------------------------------------------------- helpers
__device__ inline float hmax16(float v) {
#pragma unroll
    for (int off = 1; off < 16; off <<= 1) v = fmaxf(v, __shfl_xor(v, off, 32));
    return v;
}
__device__ inline float hsum16(float v) {
#pragma unroll
    for (int off = 1; off < 16; off <<= 1) v += __shfl_xor(v, off, 32);
    return v;
}

// ---------------------------------------------------------------- fused causal flash attention
// One wave (32 threads) per 16-row Q tile of one (batch, head).
// qh: [MTOT, DIM] f16 (pre-scaled by 1/sqrt(D)); kh: [MTOT, DH]; vT: [DH, MTOT]
__global__ __launch_bounds__(32)
void attn_fused(const _Float16* __restrict__ qh, const _Float16* __restrict__ kh,
                const _Float16* __restrict__ vT, const unsigned char* __restrict__ mask,
                _Float16* __restrict__ oh) {
    __shared__ __align__(32) _Float16 lp[16 * 32];   // P tile staging (16x32)

    const int lane = threadIdx.x;
    const int l15  = lane & 15;
    const int hi   = lane >> 4;
    const int ak0  = hi ? 16 : 0;
    const int bk0  = hi ? 16 : 0;
    const int rt = blockIdx.x;           // row tile 0..SEQ/16-1
    const int h  = blockIdx.y;
    const int b  = blockIdx.z;
    const int i0 = rt * 16;
    const int tb = b * SEQ;

    // Q A-fragments (16x64 -> two 16x32 chunks), held in registers
    const _Float16* qrow = qh + (size_t)(tb + i0 + l15) * DIM + h * DH;
    v16h aq[2];
#pragma unroll
    for (int c = 0; c < 2; ++c) {
        v8h lo = *(const v8h*)(qrow + 32 * c + ak0);
        v8h hh = *(const v8h*)(qrow + 32 * c + 8 + ak0);
#pragma unroll
        for (int i = 0; i < 8; ++i) { aq[c][i] = lo[i]; aq[c][i + 8] = hh[i]; }
    }

    float mrow[8], lrow[8];
    bool  qm[8];
    v8f   o[4] = {};
#pragma unroll
    for (int r = 0; r < 8; ++r) {
        mrow[r] = NEG_MAX; lrow[r] = 0.0f;
        qm[r] = mask[(size_t)b * SEQ + i0 + r + 8 * hi] != 0;
    }

    for (int jb = 0; jb < i0 + 16; jb += 32) {
        // S = Q @ K^T for 32 keys -> two 16x16 score tiles
        v8f s0 = {}, s1 = {};
#pragma unroll
        for (int c = 0; c < 2; ++c) {
            v16h bkA = *(const v16h*)(kh + (size_t)(tb + jb + l15) * DH + 32 * c + bk0);
            s0 = __builtin_amdgcn_wmma_f32_16x16x32_f16(false, aq[c], false, bkA, (short)0, s0, false, false);
            v16h bkB = *(const v16h*)(kh + (size_t)(tb + jb + 16 + l15) * DH + 32 * c + bk0);
            s1 = __builtin_amdgcn_wmma_f32_16x16x32_f16(false, aq[c], false, bkB, (short)0, s1, false, false);
        }

        // causal + query mask, online softmax
        float p0[8], p1[8], alpha[8];
#pragma unroll
        for (int r = 0; r < 8; ++r) {
            const int i = i0 + r + 8 * hi;
            const int j = jb + l15;
            float e0 = (j > i      || !qm[r]) ? NEG_MAX : s0[r];
            float e1 = (j + 16 > i || !qm[r]) ? NEG_MAX : s1[r];
            float rm = hmax16(fmaxf(e0, e1));
            float nm = fmaxf(mrow[r], rm);
            alpha[r] = __expf(mrow[r] - nm);
            p0[r] = __expf(e0 - nm);
            p1[r] = __expf(e1 - nm);
            float rs = hsum16(p0[r] + p1[r]);
            lrow[r] = lrow[r] * alpha[r] + rs;
            mrow[r] = nm;
        }
#pragma unroll
        for (int t = 0; t < 4; ++t)
#pragma unroll
            for (int r = 0; r < 8; ++r) o[t][r] *= alpha[r];

        // stage P (16x32, f16) through LDS to re-layout into an A fragment
#pragma unroll
        for (int r = 0; r < 8; ++r) {
            const int mloc = r + 8 * hi;
            lp[mloc * 32 + l15]      = (_Float16)p0[r];
            lp[mloc * 32 + 16 + l15] = (_Float16)p1[r];
        }
        __syncthreads();
        v16h ap;
        {
            const _Float16* prow = lp + l15 * 32;
            v8h lo = *(const v8h*)(prow + ak0);
            v8h hh = *(const v8h*)(prow + 8 + ak0);
#pragma unroll
            for (int i = 0; i < 8; ++i) { ap[i] = lo[i]; ap[i + 8] = hh[i]; }
        }
        __syncthreads();

        // O += P @ V (V^T is dh-major: contiguous B-fragment loads)
#pragma unroll
        for (int t = 0; t < 4; ++t) {
            v16h bv = *(const v16h*)(vT + (size_t)(t * 16 + l15) * MTOT + tb + jb + bk0);
            o[t] = __builtin_amdgcn_wmma_f32_16x16x32_f16(false, ap, false, bv, (short)0, o[t], false, false);
        }
    }

    // normalize and write
#pragma unroll
    for (int r = 0; r < 8; ++r) {
        const float inv = 1.0f / lrow[r];
        _Float16* orow = oh + (size_t)(tb + i0 + r + 8 * hi) * DIM + h * DH;
#pragma unroll
        for (int t = 0; t < 4; ++t) orow[t * 16 + l15] = (_Float16)(o[t][r] * inv);
    }
}

// ---------------------------------------------------------------- launch
extern "C" void kernel_launch(void* const* d_in, const int* in_sizes, int n_in,
                              void* d_out, int out_size, void* d_ws, size_t ws_size,
                              hipStream_t stream) {
    const float*         x    = (const float*)d_in[0];
    const unsigned char* mask = (const unsigned char*)d_in[1];
    const float*         Wq   = (const float*)d_in[2];
    const float*         Wk   = (const float*)d_in[3];
    const float*         Wv   = (const float*)d_in[4];
    const float*         Wfc  = (const float*)d_in[5];
    const float*         bfc  = (const float*)d_in[6];
    float*               out  = (float*)d_out;

    char* ws = (char*)d_ws;
    size_t off = 0;
    auto alloc = [&](size_t bytes) -> char* {
        char* p = ws + off;
        off += (bytes + 255) & ~(size_t)255;
        return p;
    };
    _Float16* xh   = (_Float16*)alloc((size_t)MTOT * DIM * 2);
    _Float16* wqh  = (_Float16*)alloc((size_t)DIM * DIM * 2);
    _Float16* wkh  = (_Float16*)alloc((size_t)DH * DIM * 2);
    _Float16* wvh  = (_Float16*)alloc((size_t)DH * DIM * 2);
    _Float16* wfch = (_Float16*)alloc((size_t)DIM * DIM * 2);
    _Float16* qh   = (_Float16*)alloc((size_t)MTOT * DIM * 2);
    _Float16* kh   = (_Float16*)alloc((size_t)MTOT * DH * 2);
    _Float16* vT   = (_Float16*)alloc((size_t)DH * MTOT * 2);
    _Float16* oh   = (_Float16*)alloc((size_t)MTOT * DIM * 2);

    f32_to_f16<<<256, 256, 0, stream>>>(x,   xh,   MTOT * DIM);
    f32_to_f16<<<256, 256, 0, stream>>>(Wq,  wqh,  DIM * DIM);
    f32_to_f16<<<64,  256, 0, stream>>>(Wk,  wkh,  DH * DIM);
    f32_to_f16<<<64,  256, 0, stream>>>(Wv,  wvh,  DH * DIM);
    f32_to_f16<<<256, 256, 0, stream>>>(Wfc, wfch, DIM * DIM);

    // Q = (x Wq^T) * scale  -> f16 [MTOT, DIM]
    gemm_nt<<<dim3(DIM / 64, MTOT / 128), 256, 0, stream>>>(
        xh, wqh, MTOT, DIM, DIM, DIM, DIM, qh, nullptr, DIM, kScale, nullptr);
    // K = x Wk^T            -> f16 [MTOT, DH]
    gemm_nt<<<dim3(1, MTOT / 128), 256, 0, stream>>>(
        xh, wkh, MTOT, DH, DIM, DIM, DIM, kh, nullptr, DH, 1.0f, nullptr);
    // V^T = Wv x^T          -> f16 [DH, MTOT]  (dh-major for the PV WMMA)
    gemm_nt<<<dim3(MTOT / 64, 1), 256, 0, stream>>>(
        wvh, xh, DH, MTOT, DIM, DIM, DIM, vT, nullptr, MTOT, 1.0f, nullptr);

    // fused causal flash attention -> f16 [MTOT, DIM]
    attn_fused<<<dim3(SEQ / 16, NH, B_SZ), 32, 0, stream>>>(qh, kh, vT, mask, oh);

    // out = O Wfc^T + bfc   -> f32
    gemm_nt<<<dim3(DIM / 64, MTOT / 128), 256, 0, stream>>>(
        oh, wfch, MTOT, DIM, DIM, DIM, DIM, nullptr, out, DIM, 1.0f, bfc);
}
